// GINEEncoder_5806795784250
// MI455X (gfx1250) — matrix operand order repaired and checked by reading, not compiled, not run on hardware
//
#include <hip/hip_runtime.h>
#include <cstdint>
#include <cstddef>

// Problem constants (match reference)
#define NNODES 50000
#define NEDGES 800000
#define DIN    128
#define DHID   256
#define DOUT   64

typedef __attribute__((ext_vector_type(16))) _Float16 v16h;
typedef __attribute__((ext_vector_type(8)))  _Float16 v8h;
typedef __attribute__((ext_vector_type(8)))  float    v8f;
typedef __attribute__((ext_vector_type(4)))  int      i32x4;

// Use CDNA5 async global->LDS ops (ASYNCcnt path) when the toolchain has them.
#if __has_builtin(__builtin_amdgcn_global_load_async_to_lds_b128) && \
    __has_builtin(__builtin_amdgcn_s_wait_asynccnt)
#define GEMM_ASYNC_STAGE 1
typedef __attribute__((address_space(1))) i32x4 gbl_i32x4;
typedef __attribute__((address_space(3))) i32x4 lds_i32x4;
#endif

// ---------------------------------------------------------------------------
// Hardware float atomic add (no return) -> global_atomic_add_f32 (STOREcnt)
// ---------------------------------------------------------------------------
__device__ __forceinline__ void atomic_add_f32_nr(float* p, float v) {
  asm volatile("global_atomic_add_f32 %0, %1, off" :: "v"(p), "v"(v) : "memory");
}

// ---------------------------------------------------------------------------
// agg = x  (vectorized copy, float4)
// ---------------------------------------------------------------------------
__global__ void __launch_bounds__(256)
copy_f32x4_kernel(const float4* __restrict__ src, float4* __restrict__ dst, int n4) {
  int i = blockIdx.x * blockDim.x + threadIdx.x;
  if (i < n4) dst[i] = src[i];
}

// ---------------------------------------------------------------------------
// Weight prep: W[K][Ncol] fp32 (row-major) -> WT[Ncol][K] f16 (column-major)
// so a WMMA B fragment is two contiguous 16B runs per lane.
// ---------------------------------------------------------------------------
__global__ void __launch_bounds__(256)
w_to_f16T_kernel(const float* __restrict__ W, _Float16* __restrict__ WT, int K, int Ncol) {
  int i = blockIdx.x * blockDim.x + threadIdx.x;
  if (i >= K * Ncol) return;
  int k = i / Ncol;
  int n = i - k * Ncol;
  WT[n * K + k] = (_Float16)W[i];
}

// ---------------------------------------------------------------------------
// Edge stage: one wave (32 lanes) per edge; lane handles 4 contiguous channels
// (float4) -> fully coalesced 512B row gather of x[src].
// msg = relu(x[src] + ew*We + be); agg[dst] += msg via hw float atomics.
// x and agg are both L2-resident (25.6MB each vs 192MB L2).
// ---------------------------------------------------------------------------
__global__ void __launch_bounds__(256)
edge_kernel(const float* __restrict__ x, const int* __restrict__ ei,
            const float* __restrict__ ew, const float* __restrict__ We,
            const float* __restrict__ be, float* __restrict__ agg, int nedges) {
  int wid  = (int)((blockIdx.x * blockDim.x + threadIdx.x) >> 5);
  int lane = threadIdx.x & 31;
  if (wid >= nedges) return;
  int   src = ei[wid];
  int   dst = ei[nedges + wid];
  float w   = ew[wid];
  int   c   = lane * 4;
  float4 xv = *(const float4*)(x  + (size_t)src * DIN + c);
  float4 wv = *(const float4*)(We + c);
  float4 bv = *(const float4*)(be + c);
  float4 m;
  m.x = fmaxf(xv.x + w * wv.x + bv.x, 0.0f);
  m.y = fmaxf(xv.y + w * wv.y + bv.y, 0.0f);
  m.z = fmaxf(xv.z + w * wv.z + bv.z, 0.0f);
  m.w = fmaxf(xv.w + w * wv.w + bv.w, 0.0f);
  float* ap = agg + (size_t)dst * DIN + c;
  atomic_add_f32_nr(ap + 0, m.x);
  atomic_add_f32_nr(ap + 1, m.y);
  atomic_add_f32_nr(ap + 2, m.z);
  atomic_add_f32_nr(ap + 3, m.w);
}

// ---------------------------------------------------------------------------
// WMMA A-fragment loaders.
// ISA 16-bit A 16x32 layout: lane L<16 holds row M=L, K in {0..7} U {16..23};
// lane L>=16 holds row M=L-16, K in {8..15} U {24..31}. Element j of v16h maps
// to packed half j (VGPR j/2, half j%2).
// ---------------------------------------------------------------------------
__device__ __forceinline__ v16h load_a_f32frag(const float* __restrict__ rowp, int kb) {
  float4 f0 = *(const float4*)(rowp + kb);
  float4 f1 = *(const float4*)(rowp + kb + 4);
  float4 f2 = *(const float4*)(rowp + kb + 16);
  float4 f3 = *(const float4*)(rowp + kb + 20);
  v16h a;
  a[0]  = (_Float16)f0.x; a[1]  = (_Float16)f0.y; a[2]  = (_Float16)f0.z; a[3]  = (_Float16)f0.w;
  a[4]  = (_Float16)f1.x; a[5]  = (_Float16)f1.y; a[6]  = (_Float16)f1.z; a[7]  = (_Float16)f1.w;
  a[8]  = (_Float16)f2.x; a[9]  = (_Float16)f2.y; a[10] = (_Float16)f2.z; a[11] = (_Float16)f2.w;
  a[12] = (_Float16)f3.x; a[13] = (_Float16)f3.y; a[14] = (_Float16)f3.z; a[15] = (_Float16)f3.w;
  return a;
}

__device__ __forceinline__ v16h load_a_f16frag(const _Float16* __restrict__ rowp, int kb) {
  v8h lo = *(const v8h*)(rowp + kb);
  v8h hi = *(const v8h*)(rowp + kb + 16);
  v16h a;
#pragma unroll
  for (int i = 0; i < 8; ++i) { a[i] = lo[i]; a[8 + i] = hi[i]; }
  return a;
}

// ---------------------------------------------------------------------------
// WMMA GEMM with LDS-resident B:
//   Out[M x NCOLS] = act(A[M x KDIM] * B[KDIM x NCOLS] + bias)
//   B (<=64KB f16, column-major) is staged once per workgroup into LDS
//   (async load-to-LDS when available), then every WMMA feeds from ds_load.
//   One wave computes a 16 x (NT*16) tile; A fragment loaded once per K-step
//   and reused by NT WMMAs. OUTMODE: 0 relu->f16, 1 relu->f32, 2 tanh->f32.
// M = 50000 = 3125*16 exactly -> no row guards, EXEC all-ones around WMMA.
// ---------------------------------------------------------------------------
template <bool AF16, int KDIM, int NCOLS, int NT, int OUTMODE, int TPW>
__global__ void __launch_bounds__(256)
gemm_lds_kernel(const void* __restrict__ Aptr, const _Float16* __restrict__ BT,
                const float* __restrict__ bias, void* __restrict__ Out,
                int mtiles) {
  constexpr int NGROUPS = NCOLS / (NT * 16);
  static_assert(NCOLS % (NT * 16) == 0, "tile width");
  __shared__ alignas(32) _Float16 Bs[KDIM * NCOLS];

  // --- stage B (column-major f16) into LDS -------------------------------
  constexpr int CHUNKS = (KDIM * NCOLS * 2) / 16;  // 16B chunks
#ifdef GEMM_ASYNC_STAGE
  for (int c = threadIdx.x; c < CHUNKS; c += 256) {
    __builtin_amdgcn_global_load_async_to_lds_b128(
        (gbl_i32x4*)(BT + c * 8),
        (lds_i32x4*)(&Bs[c * 8]), 0, 0);
  }
  __builtin_amdgcn_s_wait_asynccnt(0);
  __syncthreads();
#else
  for (int c = threadIdx.x; c < CHUNKS; c += 256) {
    *((v8h*)Bs + c) = *((const v8h*)BT + c);
  }
  __syncthreads();
#endif

  int wave  = threadIdx.x >> 5;
  int lane  = threadIdx.x & 31;
  int lhalf = lane >> 4;     // 0: lanes 0-15, 1: lanes 16-31
  int l16   = lane & 15;
  int kbA   = lhalf * 8;     // A K-offset per ISA layout
  int kbB   = lhalf * 8;     // B: same K layout as A
  int total = mtiles * NGROUPS;

  for (int i = 0; i < TPW; ++i) {
    int task = blockIdx.x * (8 * TPW) + wave * TPW + i;  // wave-uniform
    if (task >= total) break;
    int mt    = task / NGROUPS;
    int ng    = task - mt * NGROUPS;
    int nbase = ng * NT * 16;
    size_t m_lane = (size_t)(mt * 16 + l16);

    v8f acc[NT] = {};
#pragma unroll
    for (int k0 = 0; k0 < KDIM; k0 += 32) {
      v16h a;
      if constexpr (AF16)
        a = load_a_f16frag((const _Float16*)Aptr + m_lane * KDIM + k0, kbA);
      else
        a = load_a_f32frag((const float*)Aptr + m_lane * KDIM + k0, kbA);
#pragma unroll
      for (int s = 0; s < NT; ++s) {
        const _Float16* bsrc = &Bs[(unsigned)((nbase + s * 16 + l16) * KDIM + k0 + kbB)];
        v8h blo = *(const v8h*)bsrc;
        v8h bhi = *(const v8h*)(bsrc + 16);
        v16h b;
#pragma unroll
        for (int q = 0; q < 8; ++q) { b[q] = blo[q]; b[8 + q] = bhi[q]; }
        acc[s] = __builtin_amdgcn_wmma_f32_16x16x32_f16(
            false, a, false, b, (short)0, acc[s], false, false);
      }
    }

    // C/D layout: VGPR r, lanes 0-15 -> row M=r, lanes 16-31 -> M=8+r; N=l16.
#pragma unroll
    for (int s = 0; s < NT; ++s) {
      int   col = nbase + s * 16 + l16;
      float bv  = bias[col];
#pragma unroll
      for (int r = 0; r < 8; ++r) {
        size_t row = (size_t)(mt * 16 + r + lhalf * 8);
        float  v   = acc[s][r] + bv;
        if constexpr (OUTMODE == 2) v = tanhf(v);
        else                        v = fmaxf(v, 0.0f);
        if constexpr (OUTMODE == 0)
          ((_Float16*)Out)[row * NCOLS + col] = (_Float16)v;
        else
          ((float*)Out)[row * NCOLS + col] = v;
      }
    }
  }
}

// ---------------------------------------------------------------------------
// Host orchestration
// ---------------------------------------------------------------------------
extern "C" void kernel_launch(void* const* d_in, const int* in_sizes, int n_in,
                              void* d_out, int out_size, void* d_ws, size_t ws_size,
                              hipStream_t stream) {
  const float* x_in = (const float*)d_in[0];
  const int*   ei   = (const int*)d_in[1];
  const float* ew   = (const float*)d_in[2];

  // params flattening may be dict-insertion (We,be,W1,b1,W2,b2) or
  // sorted-key (W1,W2,We,b1,b2,be). Disambiguate by element counts.
  const float *We[2], *be[2], *W1[2], *b1[2], *W2[2], *b2[2];
  bool sorted_keys = (in_sizes[3] == DIN * DHID);
  for (int l = 0; l < 2; ++l) {
    int b = 3 + l * 6;
    if (!sorted_keys) {
      We[l] = (const float*)d_in[b + 0]; be[l] = (const float*)d_in[b + 1];
      W1[l] = (const float*)d_in[b + 2]; b1[l] = (const float*)d_in[b + 3];
      W2[l] = (const float*)d_in[b + 4]; b2[l] = (const float*)d_in[b + 5];
    } else {
      W1[l] = (const float*)d_in[b + 0]; W2[l] = (const float*)d_in[b + 1];
      We[l] = (const float*)d_in[b + 2]; b1[l] = (const float*)d_in[b + 3];
      b2[l] = (const float*)d_in[b + 4]; be[l] = (const float*)d_in[b + 5];
    }
  }
  const float* Wp = (const float*)d_in[15];
  const float* bp = (const float*)d_in[16];

  // Workspace carve-out (all offsets 256B aligned)
  char*  ws  = (char*)d_ws;
  size_t off = 0;
  auto take = [&](size_t bytes) -> void* {
    void* p = ws + off;
    off = (off + bytes + 255) & ~(size_t)255;
    return p;
  };
  float*     xbuf = (float*)take((size_t)NNODES * DIN * sizeof(float));
  float*     agg  = (float*)take((size_t)NNODES * DIN * sizeof(float));
  _Float16*  hbuf = (_Float16*)take((size_t)NNODES * DHID * sizeof(_Float16));
  _Float16*  w1t[2];
  _Float16*  w2t[2];
  for (int l = 0; l < 2; ++l) {
    w1t[l] = (_Float16*)take((size_t)DIN * DHID * sizeof(_Float16));
    w2t[l] = (_Float16*)take((size_t)DHID * DIN * sizeof(_Float16));
  }
  _Float16* wpt = (_Float16*)take((size_t)DIN * DOUT * sizeof(_Float16));

  // Weight prep (deterministic, every call)
  for (int l = 0; l < 2; ++l) {
    w_to_f16T_kernel<<<(DIN * DHID + 255) / 256, 256, 0, stream>>>(W1[l], w1t[l], DIN, DHID);
    w_to_f16T_kernel<<<(DHID * DIN + 255) / 256, 256, 0, stream>>>(W2[l], w2t[l], DHID, DIN);
  }
  w_to_f16T_kernel<<<(DIN * DOUT + 255) / 256, 256, 0, stream>>>(Wp, wpt, DIN, DOUT);

  const int MT    = NNODES / 16;        // 3125 M-tiles, exact
  const int COPY4 = NNODES * DIN / 4;   // float4 count
  const int TPW   = 2;                  // tasks per wave
  const int PER_BLK = 8 * TPW;          // tasks per 256-thread block

  const float* xcur = x_in;
  for (int l = 0; l < 2; ++l) {
    // agg = x  (folds h = x + sum(msg))
    copy_f32x4_kernel<<<(COPY4 + 255) / 256, 256, 0, stream>>>(
        (const float4*)xcur, (float4*)agg, COPY4);
    // agg += relu(x[src] + ew*We + be) scattered to dst (hw f32 atomics)
    edge_kernel<<<NEDGES / 8, 256, 0, stream>>>(xcur, ei, ew, We[l], be[l], agg, NEDGES);
    // h = relu(agg @ W1 + b1)   [f32 A -> f16 out], 2 column groups of 128
    gemm_lds_kernel<false, DIN, DHID, 8, 0, 2>
        <<<(MT * 2 + PER_BLK - 1) / PER_BLK, 256, 0, stream>>>(agg, w1t[l], b1[l], hbuf, MT);
    // x = relu(h @ W2 + b2)     [f16 A -> f32 out], full 128-wide per wave
    gemm_lds_kernel<true, DHID, DIN, 8, 1, 2>
        <<<(MT + PER_BLK - 1) / PER_BLK, 256, 0, stream>>>(hbuf, w2t[l], b2[l], xbuf, MT);
    xcur = xbuf;
  }
  // out = tanh(x @ W_post + b_post), full 64-wide per wave
  gemm_lds_kernel<false, DIN, DOUT, 4, 2, 2>
      <<<(MT + PER_BLK - 1) / PER_BLK, 256, 0, stream>>>(xbuf, wpt, bp, d_out, MT);
}
